// CAML_19756849561644
// MI455X (gfx1250) — matrix operand order repaired and checked by reading, not compiled, not run on hardware
//
#include <hip/hip_runtime.h>

typedef __bf16 bf16;
typedef __attribute__((ext_vector_type(16))) __bf16 v16bf;
typedef __attribute__((ext_vector_type(8)))  __bf16 v8bf;
typedef __attribute__((ext_vector_type(8)))  float  v8f;

constexpr int B_  = 8;
constexpr int S_  = 2048;
constexpr int D_  = 512;    // d_model
constexpr int K_  = 256;    // n_kernels
constexpr int KS_ = 9;
constexpr int C_  = 4096;
constexpr int PAD_ = 4;

// att_k LDS staging: chunk of SCHUNK s-rows, padded row stride to kill bank conflicts.
constexpr int SCHUNK = 32;          // rows per staged chunk
constexpr int LROW   = 264;         // bf16 elems per LDS row (132 dwords -> lane m on bank 4m)
constexpr int NCHUNK = S_ / SCHUNK; // 64

// Workspace layout (bytes)
constexpr size_t OFF_X1 = 0;                                   // B*S*D bf16
constexpr size_t OFF_X2 = OFF_X1 + (size_t)B_*S_*D_*2;
constexpr size_t OFF_WB = OFF_X2 + (size_t)B_*S_*K_*2;
constexpr size_t OFF_UA = OFF_WB + (size_t)KS_*D_*K_*2;
constexpr size_t OFF_FA = OFF_UA + (size_t)C_*K_*2;

#if defined(__HIP_DEVICE_COMPILE__) && __has_builtin(__builtin_amdgcn_global_load_async_to_lds_b128)
#define HAVE_ASYNC_LDS 1
#else
#define HAVE_ASYNC_LDS 0
#endif

#if HAVE_ASYNC_LDS
typedef int v4i_ __attribute__((ext_vector_type(4)));
typedef __attribute__((address_space(1))) v4i_* as1_v4i;   // global ptr (prints as __device__)
typedef __attribute__((address_space(3))) v4i_* as3_v4i;   // LDS ptr    (prints as __shared__)
#if __has_builtin(__builtin_amdgcn_s_wait_asynccnt)
#define ASYNC_WAIT() __builtin_amdgcn_s_wait_asynccnt(0)
#else
#define ASYNC_WAIT() asm volatile("s_wait_asynccnt 0" ::: "memory")
#endif
#else
#define ASYNC_WAIT()
#endif

__device__ __forceinline__ v16bf ld16(const bf16* p) {
    v8bf lo = *(const v8bf*)p;
    v8bf hi = *(const v8bf*)(p + 8);
    return __builtin_shufflevector(lo, hi, 0,1,2,3,4,5,6,7,8,9,10,11,12,13,14,15);
}

// Pre-swizzle conv weights [K][D][KS] f32 -> B-fragment layout, bf16.
// Fragment f = (t*16 + dc)*16 + kt holds the 32(d) x 16(k) tile for tap t.
// B-layout: element(d_local, n): lane = n + 16*((d_local>>3)&1), h = (d_local&7) + 8*(d_local>=16).
__global__ void swz_convw_k(const float* __restrict__ w, bf16* __restrict__ wB) {
    int idx = blockIdx.x * 256 + threadIdx.x;      // < KS*D*K = 1179648
    int f = idx >> 9, e = idx & 511;
    int lane = e >> 4, h = e & 15;
    int kt = f & 15, dc = (f >> 4) & 15, t = f >> 8;
    int n  = lane & 15;
    int dl = (h & 7) + ((lane >> 4) << 3) + ((h >> 3) << 4);
    int k  = kt * 16 + n;
    int d  = dc * 32 + dl;
    wB[idx] = (bf16)w[((size_t)k * D_ + d) * KS_ + t];
}

// Pre-swizzle U_w and final_w [C][K] f32 -> A-fragment layout, bf16.
// A-layout: element(m, K): lane&15 = m, K = h + 16*(lane>=16).
__global__ void swz_att_k(const float* __restrict__ U, const float* __restrict__ F,
                          bf16* __restrict__ UA, bf16* __restrict__ FA) {
    int idx = blockIdx.x * 256 + threadIdx.x;      // < C*K = 1048576
    int f = idx >> 9, e = idx & 511;
    int lane = e >> 4, h = e & 15;
    int ct = f >> 3, kc = f & 7;
    int m  = lane & 15, hi = lane >> 4;
    int kl = h + 16 * hi;
    size_t src = (size_t)(ct * 16 + m) * K_ + kc * 32 + kl;
    UA[idx] = (bf16)U[src];
    FA[idx] = (bf16)F[src];
}

// Embedding gather -> bf16 x1 [B*S][D], coalesced along D.
__global__ void embed_k(const int* __restrict__ text, const float* __restrict__ table,
                        bf16* __restrict__ x1) {
    size_t idx = (size_t)blockIdx.x * 256 + threadIdx.x;   // < B*S*D
    int d = (int)(idx & (D_ - 1));
    int r = (int)(idx >> 9);                               // D = 512
    int tok = text[r];
    x1[idx] = (bf16)table[(size_t)tok * D_ + d];
}

// Conv1d (implicit GEMM) + bias + tanh -> bf16 x2 [B*S][K].
// Block = 8 waves; block owns one 16-row s-tile, wave w owns k-tiles 2w, 2w+1.
__global__ void __launch_bounds__(256) conv_k(const bf16* __restrict__ x1,
                                              const bf16* __restrict__ wB,
                                              const float* __restrict__ bias,
                                              bf16* __restrict__ x2) {
    int g    = blockIdx.x;            // s-tile index, [0, B*S/16)
    int lane = threadIdx.x & 31;
    int wv   = threadIdx.x >> 5;
    int m    = lane & 15, hi = lane >> 4;
    int b    = (g * 16) / S_;
    int sl   = (g * 16) % S_;         // tile base inside the batch
    int kt0  = 2 * wv, kt1 = 2 * wv + 1;

    v8f acc0 = {}, acc1 = {};

    for (int t = 0; t < KS_; ++t) {
        int srow   = sl + m + t - PAD_;              // per-lane input row
        bool valid = (srow >= 0) && (srow < S_);
        const bf16* xrow = x1 + ((size_t)b * S_ + srow) * D_;
        for (int dc = 0; dc < 16; ++dc) {
            int d0 = dc * 32 + 8 * hi;
            v16bf a = {};
            if (valid) {
                v8bf lo  = *(const v8bf*)(xrow + d0);
                v8bf hi8 = *(const v8bf*)(xrow + d0 + 16);
                a = __builtin_shufflevector(lo, hi8, 0,1,2,3,4,5,6,7,8,9,10,11,12,13,14,15);
            }
            const bf16* bp = wB + ((size_t)((t * 16 + dc) * 16)) * 512;
            v16bf b0 = ld16(bp + (size_t)kt0 * 512 + lane * 16);
            v16bf b1 = ld16(bp + (size_t)kt1 * 512 + lane * 16);
            acc0 = __builtin_amdgcn_wmma_f32_16x16x32_bf16(false, a, false, b0,
                                                           (short)0, acc0, false, false);
            acc1 = __builtin_amdgcn_wmma_f32_16x16x32_bf16(false, a, false, b1,
                                                           (short)0, acc1, false, false);
        }
    }

    int kg0 = kt0 * 16 + m, kg1 = kt1 * 16 + m;
    float bi0 = bias[kg0], bi1 = bias[kg1];
#pragma unroll
    for (int i = 0; i < 8; ++i) {
        size_t row = (size_t)g * 16 + i + 8 * hi;    // absolute row in [0, B*S)
        x2[row * K_ + kg0] = (bf16)tanhf(acc0[i] + bi0);
        x2[row * K_ + kg1] = (bf16)tanhf(acc1[i] + bi1);
    }
}

// Stage one SCHUNK-row chunk of x2[b] (row-major, 256 bf16/row) into LDS with
// padded rows. 1024 x b128 transfers, 4 per thread, via async global->LDS DMA.
__device__ __forceinline__ void stage_chunk(const bf16* __restrict__ src, bf16* dst) {
#pragma unroll
    for (int i = 0; i < 4; ++i) {
        int j   = threadIdx.x + i * 256;   // < SCHUNK*32
        int row = j >> 5;
        int col = j & 31;                  // 32 x 16B per row
        const bf16* gp = src + (size_t)row * K_ + col * 8;
        bf16*       lp = dst + row * LROW + col * 8;
#if HAVE_ASYNC_LDS
        __builtin_amdgcn_global_load_async_to_lds_b128(
            (as1_v4i)(unsigned long long)gp,
            (as3_v4i)(unsigned int)(unsigned long long)lp, 0, 0);
#else
        *(uint4*)lp = *(const uint4*)gp;
#endif
    }
}

// Fused label attention: score GEMM + online softmax + value GEMM + pooling.
// Grid = B * (C/128); block = 8 waves; wave owns 16 classes. The x2 s-chunks
// are staged once per block into double-buffered LDS via async copies, so the
// 8 waves share one copy instead of 8 redundant global streams.
__global__ void __launch_bounds__(256) att_k(const bf16* __restrict__ x2,
                                             const bf16* __restrict__ UA,
                                             const bf16* __restrict__ FA,
                                             const float* __restrict__ fb,
                                             float* __restrict__ out) {
    __shared__ bf16 lbuf[2][SCHUNK * LROW];   // 2 x 16.5 KB

    int b    = blockIdx.x >> 5;
    int cb   = blockIdx.x & 31;
    int wv   = threadIdx.x >> 5;
    int lane = threadIdx.x & 31;
    int ct   = cb * 8 + wv;            // c-tile index, [0, 256)
    int m    = lane & 15, hi = lane >> 4;

    const bf16* xg = x2 + (size_t)b * S_ * K_;
    stage_chunk(xg, lbuf[0]);          // prologue copy overlaps fragment preload

    // Persistent A fragments for this wave's 16 classes (U rows + final_w rows).
    v16bf ua[8], fa[8];
#pragma unroll
    for (int kc = 0; kc < 8; ++kc) {
        ua[kc] = ld16(UA + ((size_t)(ct * 8 + kc)) * 512 + lane * 16);
        fa[kc] = ld16(FA + ((size_t)(ct * 8 + kc)) * 512 + lane * 16);
    }

    float sume[8] = {0.f,0.f,0.f,0.f,0.f,0.f,0.f,0.f};
    float sacc[8] = {0.f,0.f,0.f,0.f,0.f,0.f,0.f,0.f};

    ASYNC_WAIT();
    __syncthreads();

    for (int c0 = 0; c0 < NCHUNK; ++c0) {
        if (c0 + 1 < NCHUNK)
            stage_chunk(xg + (size_t)(c0 + 1) * SCHUNK * K_, lbuf[(c0 + 1) & 1]);

        const bf16* lbase = lbuf[c0 & 1] + m * LROW + 16 * hi;
#pragma unroll
        for (int sub = 0; sub < SCHUNK / 16; ++sub) {
            const bf16* lp = lbase + sub * 16 * LROW;
            v8f sc = {}, vv = {};
#pragma unroll
            for (int kc = 0; kc < 8; ++kc) {
                v16bf xb = ld16(lp + kc * 32);   // ds_load_b128 pair, bank-padded
                sc = __builtin_amdgcn_wmma_f32_16x16x32_bf16(false, ua[kc], false, xb,
                                                             (short)0, sc, false, false);
                vv = __builtin_amdgcn_wmma_f32_16x16x32_bf16(false, fa[kc], false, xb,
                                                             (short)0, vv, false, false);
            }
            // C layout: VGPR i -> class (i + 8*hi), lane&15 -> s position.
#pragma unroll
            for (int i = 0; i < 8; ++i) {
                float e = __expf(sc[i]);         // scores bounded; no max needed
                sume[i] += e;
                sacc[i] += e * vv[i];
            }
        }
        ASYNC_WAIT();      // own next-chunk copies landed (overlapped with compute)
        __syncthreads();   // everyone done reading current buffer + sees new data
    }

    // Reduce over the 16 s-lanes inside each half of the wave.
#pragma unroll
    for (int i = 0; i < 8; ++i) {
#pragma unroll
        for (int off = 1; off < 16; off <<= 1) {
            sume[i] += __shfl_xor(sume[i], off, 32);
            sacc[i] += __shfl_xor(sacc[i], off, 32);
        }
    }
    if (m == 0) {
#pragma unroll
        for (int i = 0; i < 8; ++i) {
            int c = ct * 16 + i + 8 * hi;
            out[(size_t)b * C_ + c] = sacc[i] / sume[i] + fb[c];
        }
    }
}

extern "C" void kernel_launch(void* const* d_in, const int* in_sizes, int n_in,
                              void* d_out, int out_size, void* d_ws, size_t ws_size,
                              hipStream_t stream) {
    const int*   text  = (const int*)d_in[0];
    const float* table = (const float*)d_in[1];
    const float* convw = (const float*)d_in[2];
    const float* convb = (const float*)d_in[3];
    const float* Uw    = (const float*)d_in[4];
    const float* Fw    = (const float*)d_in[5];
    const float* fb    = (const float*)d_in[6];
    float* out = (float*)d_out;

    char* ws = (char*)d_ws;
    bf16* x1 = (bf16*)(ws + OFF_X1);
    bf16* x2 = (bf16*)(ws + OFF_X2);
    bf16* wB = (bf16*)(ws + OFF_WB);
    bf16* UA = (bf16*)(ws + OFF_UA);
    bf16* FA = (bf16*)(ws + OFF_FA);

    swz_convw_k<<<(KS_*D_*K_)/256, 256, 0, stream>>>(convw, wB);
    swz_att_k  <<<(C_*K_)/256,     256, 0, stream>>>(Uw, Fw, UA, FA);
    embed_k    <<<(B_*S_*D_)/256,  256, 0, stream>>>(text, table, x1);
    conv_k     <<<(B_*S_)/16,      256, 0, stream>>>(x1, wB, convb, x2);
    att_k      <<<B_*(C_/128),     256, 0, stream>>>(x2, UA, FA, fb, out);
}
